// EigenSpace_2style_38766374814280
// MI455X (gfx1250) — compile-verified, matrix operand order checked by reading, not run on hardware
//
#include <hip/hip_runtime.h>

typedef __attribute__((ext_vector_type(16))) _Float16 v16h;
typedef __attribute__((ext_vector_type(8)))  float    v8f;

#define RES      64
#define CHANNEL  64
#define DIM      512
#define BATCH    32
#define K_TOTAL  (RES * RES * CHANNEL)        // 262144 rows to reduce
#define TILE_ROWS 32                          // K of one wmma_f32_16x16x32_f16
#define LDS_STRIDE 528                        // 512 + 16 floats pad (conflict-free column reads)
#define LDS_BUF_FLOATS (TILE_ROWS * LDS_STRIDE)
#define LDS_BUF_BYTES  (LDS_BUF_FLOATS * 4)   // 67584 B per buffer, x2 = 135168 B
#define ROWS_PER_BLOCK 256                    // 8 tiles per block
#define TILES_PER_BLOCK (ROWS_PER_BLOCK / TILE_ROWS)
#define NBLOCKS  (K_TOTAL / ROWS_PER_BLOCK)   // 1024
#define NREP     16                           // replicated partial-S arrays in d_ws (32KB)

// --- kernel 1: zero the partial-sum scratch (harness poisons d_ws) ---------
__global__ void zero_ws_kernel(float* __restrict__ ws) {
    int i = blockIdx.x * blockDim.x + threadIdx.x;
    if (i < NREP * DIM) ws[i] = 0.0f;
}

// --- kernel 2: column-sum of U via WMMA (ones^T x U) -----------------------
// Double-buffered pipeline: tile t+1 streams HBM->LDS via
// global_load_async_to_lds_b128 (ASYNCcnt) while tile t feeds
// v_wmma_f32_16x16x32_f16 with A = ones (f32 accumulate through C).
__global__ __launch_bounds__(256) void colsum_wmma_kernel(
        const float* __restrict__ U, float* __restrict__ Spart) {
    extern __shared__ float lds[];            // 2 x LDS_BUF_FLOATS

    const int t    = threadIdx.x;
    const int wave = t >> 5;
    const int lane = t & 31;
    const int ln   = lane & 15;               // lane -> column within group (N = lane&15)
    const int half = lane >> 4;               // lane-half -> even/odd row interleave

    const long rowBase = (long)blockIdx.x * ROWS_PER_BLOCK;

    // Per-thread staging coordinates: 16 float4 chunks per 32x512 tile.
    // chunk p: flat = p*256 + t -> row = flat/128, c4 = flat%128 (coalesced b128).
    auto prefetch = [&](int tl, int buf) {
#pragma unroll
        for (int p = 0; p < 16; ++p) {
            const int flat = p * 256 + t;
            const int row  = flat >> 7;
            const int c4   = flat & 127;
            const unsigned lds_off =
                (unsigned)(buf * LDS_BUF_BYTES + (row * LDS_STRIDE + c4 * 4) * 4);
            const float* g = U + (rowBase + (long)tl * TILE_ROWS + row) * DIM + c4 * 4;
            asm volatile("global_load_async_to_lds_b128 %0, %1, off"
                         :: "v"(lds_off), "v"(g)
                         : "memory");
        }
    };

    v16h ones;
#pragma unroll
    for (int i = 0; i < 16; ++i) ones[i] = (_Float16)1.0f;

    v8f acc[4] = {};                          // one accumulator per owned column-group

    prefetch(0, 0);

    for (int tl = 0; tl < TILES_PER_BLOCK; ++tl) {
        const int cur = tl & 1;
        if (tl + 1 < TILES_PER_BLOCK) {
            // Buffer cur^1 was last read in iteration tl-1; the trailing
            // barrier of that iteration makes this overwrite safe.
            prefetch(tl + 1, cur ^ 1);
            // Async loads complete in issue order per wave: allowing 16
            // outstanding (= the prefetch just issued) proves tile tl landed.
            asm volatile("s_wait_asynccnt 0x10" ::: "memory");
        } else {
            asm volatile("s_wait_asynccnt 0x0" ::: "memory");
        }
        __syncthreads();                      // all waves' tile-tl data visible

        const float* tile = lds + cur * LDS_BUF_FLOATS;
        // B operand: lane holds 16 f16 values of column d0+ln; half 0 reads
        // even rows, half 1 odd rows (A==ones makes k-slot permutation moot).
#pragma unroll
        for (int q = 0; q < 4; ++q) {
            const int d0 = (wave * 4 + q) * 16;
            v16h b;
#pragma unroll
            for (int j = 0; j < 16; ++j)
                b[j] = (_Float16)tile[(2 * j + half) * LDS_STRIDE + d0 + ln];
            acc[q] = __builtin_amdgcn_wmma_f32_16x16x32_f16(
                         /*neg_a=*/false, ones, /*neg_b=*/false, b,
                         /*c_mod=*/(short)0, acc[q],
                         /*reuse_a=*/false, /*reuse_b=*/false);
        }
        __syncthreads();                      // done reading before buffer reuse
    }

    // D rows are identical (A==ones): lanes 0-15, VGPR0 = colsum(d0+ln).
    // One atomic per column per block into a replicated partial array.
    const int rep = blockIdx.x & (NREP - 1);
    if (half == 0) {
#pragma unroll
        for (int q = 0; q < 4; ++q) {
            const int d0 = (wave * 4 + q) * 16;
            atomicAdd(&Spart[rep * DIM + d0 + ln], acc[q][0]);
        }
    }
}

// --- kernel 3: fold replicas + elementwise epilogue ------------------------
__global__ void finalize_kernel(const float* __restrict__ style,
                                const float* __restrict__ L,
                                const float* __restrict__ mu,
                                const float* __restrict__ Spart,
                                float* __restrict__ out) {
    int i = blockIdx.x * blockDim.x + threadIdx.x;
    if (i >= BATCH * DIM) return;
    int d = i & (DIM - 1);
    float s = 0.0f;
#pragma unroll
    for (int r = 0; r < NREP; ++r) s += Spart[r * DIM + d];
    out[i] = s * L[d] * style[i] + mu[d];
}

extern "C" void kernel_launch(void* const* d_in, const int* in_sizes, int n_in,
                              void* d_out, int out_size, void* d_ws, size_t ws_size,
                              hipStream_t stream) {
    const float* style = (const float*)d_in[0];   // (32, 512)
    const float* U     = (const float*)d_in[1];   // (64, 64, 64, 512)
    const float* L     = (const float*)d_in[2];   // (512,)
    const float* mu    = (const float*)d_in[3];   // (512,)
    float* out   = (float*)d_out;                 // (32, 512)
    float* Spart = (float*)d_ws;                  // NREP * 512 floats

    zero_ws_kernel<<<(NREP * DIM + 255) / 256, 256, 0, stream>>>(Spart);
    colsum_wmma_kernel<<<NBLOCKS, 256, 2 * LDS_BUF_BYTES, stream>>>(U, Spart);
    finalize_kernel<<<(BATCH * DIM + 255) / 256, 256, 0, stream>>>(style, L, mu, Spart, out);
}
